// MultiHeadSelfAttention_30468497998159
// MI455X (gfx1250) — matrix operand order repaired and checked by reading, not compiled, not run on hardware
//
#include <hip/hip_runtime.h>
#include <hip/hip_bf16.h>

// MI455X / gfx1250: wave32, WMMA 16x16x32 bf16 (f32 accum).
// Pipeline: f32->bf16 convert; QKV WMMA GEMMs with TDM (tensor_load_to_lds)
// double-buffered weight staging in LDS (V stored head-transposed);
// flash-attention with online softmax + LDS P-relayout; MLP WMMA GEMMs.

typedef __bf16 bf16;
typedef __attribute__((ext_vector_type(16))) __bf16 bf16x16;
typedef __attribute__((ext_vector_type(8)))  __bf16 bf16x8;
typedef __attribute__((ext_vector_type(8)))  float  f32x8;
typedef __attribute__((ext_vector_type(4)))  unsigned int u32x4;
typedef __attribute__((ext_vector_type(8)))  int i32x8;
typedef __attribute__((ext_vector_type(4)))  int i32x4;

#if __has_builtin(__builtin_amdgcn_tensor_load_to_lds)
#define HAVE_TDM 1
#endif

#define WMMA_BF16(a, b, c) \
  __builtin_amdgcn_wmma_f32_16x16x32_bf16(false, (a), false, (b), (short)0, (c), false, false)

static __device__ inline f32x8 zero8() {
  f32x8 z;
#pragma unroll
  for (int i = 0; i < 8; ++i) z[i] = 0.0f;
  return z;
}

// A fragment (16x32 bf16, MxK). base -> (row0, k0), row-major, lda elements.
// Lanes 0-15: row m, K = 0..7 (v0..3) and 16..23 (v4..7); lanes 16-31: K = 8..15, 24..31.
static __device__ inline bf16x16 load_a_frag(const bf16* base, int lda, int lane) {
  const int m = lane & 15, half = lane >> 4;
  const bf16* p = base + (size_t)m * lda + half * 8;
  bf16x8 lo = *(const bf16x8*)p;          // K = half*8 + 0..7
  bf16x8 hi = *(const bf16x8*)(p + 16);   // K = 16 + half*8 + 0..7
  bf16x16 r;
#pragma unroll
  for (int i = 0; i < 8; ++i) { r[i] = lo[i]; r[i + 8] = hi[i]; }
  return r;
}

// B fragment (32x16 bf16, KxN), column-major source = rows of W (NT gemm).
// base -> (col0-th row of W, k0). Lane n = column n; lane half selects K 0..7,16..23 / 8..15,24..31.
static __device__ inline bf16x16 load_b_frag(const bf16* base, int ldb, int lane) {
  const int n = lane & 15, kh = lane >> 4;
  const bf16* p = base + (size_t)n * ldb + kh * 8;
  bf16x8 lo = *(const bf16x8*)p;          // K = kh*8 + 0..7
  bf16x8 hi = *(const bf16x8*)(p + 16);   // K = 16 + kh*8 + 0..7
  bf16x16 r;
#pragma unroll
  for (int i = 0; i < 8; ++i) { r[i] = lo[i]; r[i + 8] = hi[i]; }
  return r;
}

// --- Tensor Data Mover: DMA a 64-row x 64-col bf16 tile (row pitch K elements in
// memory) into LDS with padding: 32 dwords (one 64-elem row) then +4 dwords pad
// -> LDS row pitch 72 elements (144 B), so the two 16-lane B-frag groups hit
// disjoint banks. D# layout per CDNA5 ISA.
static __device__ inline void tdm_issue_tile64(const bf16* gsrc, unsigned lds_byte_off,
                                               int K, int rows_avail) {
#if defined(HAVE_TDM)
  unsigned long long ga = (unsigned long long)(uintptr_t)gsrc;
  u32x4 g0;
  g0[0] = 1u;                                               // count=1, user descriptor
  g0[1] = lds_byte_off;                                     // lds_addr
  g0[2] = (unsigned)ga;                                     // global_addr[31:0]
  g0[3] = ((unsigned)(ga >> 32) & 0x01FFFFFFu) | 0x80000000u; // addr[56:32] | type=2
  i32x8 g1;
  g1[0] = (1 << 16)      // data_size = 2 bytes
        | (1 << 20)      // pad_enable
        | (4 << 22)      // pad_interval: every 32 dwords
        | (3 << 25);     // pad_amount: 4 dwords
  g1[1] = (K & 0xffff) << 16;                               // tensor_dim0[15:0]
  g1[2] = ((unsigned)K >> 16) | ((rows_avail & 0xffff) << 16); // td0[31:16] | td1[15:0]
  g1[3] = (((unsigned)rows_avail >> 16) & 0xffff) | (64u << 16); // td1[31:16] | tile_dim0=64
  g1[4] = 64;                                               // tile_dim1=64, tile_dim2=0
  g1[5] = K;                                                // tensor_dim0_stride[31:0]
  g1[6] = 0;
  g1[7] = 0;
  i32x4 z4 = {0, 0, 0, 0};
#if __clang_major__ >= 23
  i32x8 z8 = {0, 0, 0, 0, 0, 0, 0, 0};
  __builtin_amdgcn_tensor_load_to_lds(g0, g1, z4, z4, z8, 0);
#else
  __builtin_amdgcn_tensor_load_to_lds(g0, g1, z4, z4, 0);
#endif
#endif
}

// Fallback cooperative stage (also keeps the host-side pass legal).
static __device__ inline void stage_fallback(const bf16* gsrc, bf16* lbuf, int K, int tid) {
#pragma unroll
  for (int s = 0; s < 4; ++s) {
    const int seg = tid * 4 + s;        // 512 segments of 8 bf16
    const int row = seg >> 3;
    const int colseg = (seg & 7) * 8;
    bf16x8 v = *(const bf16x8*)(gsrc + (size_t)row * K + colseg);
    *(bf16x8*)(lbuf + row * 72 + colseg) = v;
  }
}

__global__ void to_bf16_kernel(const float* __restrict__ in, bf16* __restrict__ out, int n) {
  int i = blockIdx.x * blockDim.x + threadIdx.x;
  if (i < n) out[i] = (bf16)in[i];
}

// C = A[M,K] * W[N,K]^T + bias.  SMODE: 0=f32 row-major, 1=bf16 row-major,
// 2=bf16 head-transposed [B,H,HD,S] (for V).  128 threads = 4 waves, 64x64 block
// tile; W chunks (64x64) double-buffered in LDS via TDM.
template <int SMODE, bool RELU>
__global__ __launch_bounds__(128) void gemm_nt(const bf16* __restrict__ A,
                                               const bf16* __restrict__ W,
                                               const float* __restrict__ bias,
                                               float* __restrict__ outF,
                                               bf16* __restrict__ outB,
                                               int M, int N, int K) {
  __shared__ __align__(16) bf16 wbuf[2][64 * 72];   // 2 x 9216 B, padded pitch 72
  const int tid = threadIdx.x;
  const int lane = tid & 31;
  const int wave = tid >> 5;
  const int bm = blockIdx.y * 64 + (wave >> 1) * 32;
  const int bn0 = blockIdx.x * 64;
  const int wn = (wave & 1) * 32;

  f32x8 acc00 = zero8(), acc01 = zero8(), acc10 = zero8(), acc11 = zero8();
  const bf16* Ap0 = A + (size_t)bm * K;
  const bf16* Ap1 = A + (size_t)(bm + 16) * K;
  const bf16* Wblk = W + (size_t)bn0 * K;
  const int nch = K >> 6;

  // Stage chunk 0 -> wbuf[0].
#if defined(HAVE_TDM)
  if (wave == 0) {
    tdm_issue_tile64(Wblk, (unsigned)(uintptr_t)&wbuf[0][0], K, N - bn0);
    __builtin_amdgcn_s_wait_tensorcnt(0);
  }
#else
  stage_fallback(Wblk, &wbuf[0][0], K, tid);
#endif
  __syncthreads();

  for (int c = 0; c < nch; ++c) {
    const int kc = c << 6;
    // Kick off next chunk's DMA while this chunk computes.
#if defined(HAVE_TDM)
    if (c + 1 < nch && wave == 0)
      tdm_issue_tile64(Wblk + kc + 64, (unsigned)(uintptr_t)&wbuf[(c + 1) & 1][0], K, N - bn0);
#else
    if (c + 1 < nch)
      stage_fallback(Wblk + kc + 64, &wbuf[(c + 1) & 1][0], K, tid);
#endif

    const bf16* lb = &wbuf[c & 1][0];
#pragma unroll
    for (int kk = 0; kk < 64; kk += 32) {
      bf16x16 a0 = load_a_frag(Ap0 + kc + kk, K, lane);
      bf16x16 a1 = load_a_frag(Ap1 + kc + kk, K, lane);
      bf16x16 b0 = load_b_frag(lb + (wn + 0)  * 72 + kk, 72, lane);
      bf16x16 b1 = load_b_frag(lb + (wn + 16) * 72 + kk, 72, lane);
      acc00 = WMMA_BF16(a0, b0, acc00);
      acc01 = WMMA_BF16(a0, b1, acc01);
      acc10 = WMMA_BF16(a1, b0, acc10);
      acc11 = WMMA_BF16(a1, b1, acc11);
    }
    if (c + 1 < nch) __builtin_prefetch(Ap0 + kc + 64, 0, 3);

    __syncthreads();                 // all waves done reading wbuf[c&1]
#if defined(HAVE_TDM)
    if (c + 1 < nch) {
      if (wave == 0) __builtin_amdgcn_s_wait_tensorcnt(0);
      __syncthreads();               // wbuf[(c+1)&1] ready for everyone
    }
#endif
  }

  // C/D layout: lanes 0-15 rows 0..7 (elem r), lanes 16-31 rows 8..15; col = lane&15.
  const int ng = lane & 15, mh = lane >> 4;
  const int bn = bn0 + wn;
#pragma unroll
  for (int i = 0; i < 2; ++i) {
#pragma unroll
    for (int j = 0; j < 2; ++j) {
      f32x8 acc = (i == 0) ? ((j == 0) ? acc00 : acc01) : ((j == 0) ? acc10 : acc11);
      const int col = bn + j * 16 + ng;
      const float bval = bias ? bias[col] : 0.0f;
#pragma unroll
      for (int r = 0; r < 8; ++r) {
        const int row = bm + i * 16 + mh * 8 + r;
        float v = acc[r] + bval;
        if (RELU) v = fmaxf(v, 0.0f);
        if (SMODE == 0) {
          outF[(size_t)row * N + col] = v;
        } else if (SMODE == 1) {
          outB[(size_t)row * N + col] = (bf16)v;
        } else {
          // V transposed per head: [B=row>>10][H=col>>6][HD=col&63][S=row&1023]
          const int bb = row >> 10, s = row & 1023;
          const int hh = col >> 6, d = col & 63;
          outB[((((size_t)bb * 16 + hh) * 64 + d) << 10) + s] = (bf16)v;
        }
      }
    }
  }
}

// Flash attention. Q,K: bf16 [B,S,D] (head h at col h*64). Vt: bf16 [B,H,HD,S].
// ctx out: bf16 [B,S,D]. 4 waves/block, one 16-row Q tile per wave, 32 keys/iter.
__global__ __launch_bounds__(128) void attn_kernel(const bf16* __restrict__ Q,
                                                   const bf16* __restrict__ Km,
                                                   const bf16* __restrict__ Vt,
                                                   const int* __restrict__ mask,
                                                   bf16* __restrict__ ctx) {
  __shared__ __align__(16) bf16 lds[4][16 * 40];  // 16x32 P tile per wave, pitch 40
  const int lane = threadIdx.x & 31;
  const int wave = threadIdx.x >> 5;
  const int h = blockIdx.y, b = blockIdx.z;
  const int q0 = blockIdx.x * 64 + wave * 16;

  const bf16* Qbase = Q + ((size_t)b * 1024 + q0) * 1024 + h * 64;
  const bf16* Kbase = Km + ((size_t)b * 1024) * 1024 + h * 64;
  const bf16* Vbase = Vt + (((size_t)b * 16 + h) * 64) * 1024;

  bf16x16 qa0 = load_a_frag(Qbase, 1024, lane);
  bf16x16 qa1 = load_a_frag(Qbase + 32, 1024, lane);

  f32x8 acc[4];
#pragma unroll
  for (int t = 0; t < 4; ++t) acc[t] = zero8();
  float mrow[8], lrow[8];
#pragma unroll
  for (int r = 0; r < 8; ++r) { mrow[r] = -3.0e38f; lrow[r] = 0.0f; }

  const int ng = lane & 15, mh = lane >> 4;
  bf16* lp = lds[wave];

  for (int kc = 0; kc < 1024; kc += 32) {
    bf16x16 kb00 = load_b_frag(Kbase + (size_t)kc * 1024, 1024, lane);
    bf16x16 kb01 = load_b_frag(Kbase + (size_t)kc * 1024 + 32, 1024, lane);
    bf16x16 kb10 = load_b_frag(Kbase + (size_t)(kc + 16) * 1024, 1024, lane);
    bf16x16 kb11 = load_b_frag(Kbase + (size_t)(kc + 16) * 1024 + 32, 1024, lane);

    f32x8 s0 = zero8(), s1 = zero8();
    s0 = WMMA_BF16(qa0, kb00, s0);
    s0 = WMMA_BF16(qa1, kb01, s0);
    s1 = WMMA_BF16(qa0, kb10, s1);
    s1 = WMMA_BF16(qa1, kb11, s1);

#pragma unroll
    for (int r = 0; r < 8; ++r) {
      const int qrow = q0 + mh * 8 + r;
      const int* mp = mask + ((size_t)b * 1024 + qrow) * 1024 + kc;
      float v0 = s0[r] * 0.125f; if (mp[ng] == 0)      v0 = -1.0e8f;
      float v1 = s1[r] * 0.125f; if (mp[16 + ng] == 0) v1 = -1.0e8f;

      float cm = fmaxf(v0, v1);
#pragma unroll
      for (int off = 8; off >= 1; off >>= 1) cm = fmaxf(cm, __shfl_xor(cm, off, 32));
      const float mn = fmaxf(mrow[r], cm);
      const float sc = __expf(mrow[r] - mn);
      const float p0 = __expf(v0 - mn);
      const float p1 = __expf(v1 - mn);
      float cs = p0 + p1;
#pragma unroll
      for (int off = 8; off >= 1; off >>= 1) cs += __shfl_xor(cs, off, 32);
      lrow[r] = lrow[r] * sc + cs;
      mrow[r] = mn;
#pragma unroll
      for (int t = 0; t < 4; ++t) acc[t][r] *= sc;

      lp[(mh * 8 + r) * 40 + ng] = (bf16)p0;
      lp[(mh * 8 + r) * 40 + 16 + ng] = (bf16)p1;
    }

    bf16x16 pa = load_a_frag(lp, 40, lane);  // P as 16x32 A fragment
#pragma unroll
    for (int t = 0; t < 4; ++t) {
      bf16x16 vb = load_b_frag(Vbase + (size_t)(t * 16) * 1024 + kc, 1024, lane);
      acc[t] = WMMA_BF16(pa, vb, acc[t]);
    }
  }

  bf16* cb = ctx + ((size_t)b * 1024 + q0) * 1024 + h * 64;
#pragma unroll
  for (int t = 0; t < 4; ++t) {
#pragma unroll
    for (int r = 0; r < 8; ++r) {
      cb[(size_t)(mh * 8 + r) * 1024 + t * 16 + ng] = (bf16)(acc[t][r] / lrow[r]);
    }
  }
}

extern "C" void kernel_launch(void* const* d_in, const int* in_sizes, int n_in,
                              void* d_out, int out_size, void* d_ws, size_t ws_size,
                              hipStream_t stream) {
  const float* x   = (const float*)d_in[0];
  const float* Wq  = (const float*)d_in[1];
  const float* bq  = (const float*)d_in[2];
  const float* Wk  = (const float*)d_in[3];
  const float* bk  = (const float*)d_in[4];
  const float* Wv  = (const float*)d_in[5];
  const float* bvv = (const float*)d_in[6];
  const float* W1  = (const float*)d_in[7];
  const float* b1  = (const float*)d_in[8];
  const float* W2  = (const float*)d_in[9];
  const float* b2  = (const float*)d_in[10];
  const int* mask  = (const int*)d_in[11];
  float* out = (float*)d_out;

  // B=4, S=1024, D=1024, H=16, HD=64, D2=2048, M=B*S=4096.
  const int NX   = 4 * 1024 * 1024;
  const int NW   = 1024 * 1024;
  const int NW12 = 2048 * 1024;

  char* w = (char*)d_ws;
  bf16* xb   = (bf16*)w;
  bf16* wqb  = (bf16*)(w + (size_t)8  * 1024 * 1024);
  bf16* wkb  = (bf16*)(w + (size_t)10 * 1024 * 1024);
  bf16* wvb  = (bf16*)(w + (size_t)12 * 1024 * 1024);
  bf16* w1b  = (bf16*)(w + (size_t)14 * 1024 * 1024);
  bf16* w2b  = (bf16*)(w + (size_t)18 * 1024 * 1024);
  bf16* Qb   = (bf16*)(w + (size_t)22 * 1024 * 1024);
  bf16* Kb   = (bf16*)(w + (size_t)30 * 1024 * 1024);
  bf16* Vtb  = (bf16*)(w + (size_t)38 * 1024 * 1024);
  bf16* ctxb = (bf16*)(w + (size_t)46 * 1024 * 1024);
  bf16* h1b  = (bf16*)(w + (size_t)22 * 1024 * 1024);  // overlays Qb/Kb (dead after attn)

  to_bf16_kernel<<<(NX + 255) / 256, 256, 0, stream>>>(x, xb, NX);
  to_bf16_kernel<<<(NW + 255) / 256, 256, 0, stream>>>(Wq, wqb, NW);
  to_bf16_kernel<<<(NW + 255) / 256, 256, 0, stream>>>(Wk, wkb, NW);
  to_bf16_kernel<<<(NW + 255) / 256, 256, 0, stream>>>(Wv, wvb, NW);
  to_bf16_kernel<<<(NW12 + 255) / 256, 256, 0, stream>>>(W1, w1b, NW12);
  to_bf16_kernel<<<(NW12 + 255) / 256, 256, 0, stream>>>(W2, w2b, NW12);

  // QKV projections: M=4096, N=1024, K=1024.
  gemm_nt<1, false><<<dim3(16, 64), 128, 0, stream>>>(xb, wqb, bq,  nullptr, Qb,  4096, 1024, 1024);
  gemm_nt<1, false><<<dim3(16, 64), 128, 0, stream>>>(xb, wkb, bk,  nullptr, Kb,  4096, 1024, 1024);
  gemm_nt<2, false><<<dim3(16, 64), 128, 0, stream>>>(xb, wvb, bvv, nullptr, Vtb, 4096, 1024, 1024);

  // Attention: grid (S/64, H, B).
  attn_kernel<<<dim3(16, 16, 4), 128, 0, stream>>>(Qb, Kb, Vtb, mask, ctxb);

  // MLP: relu(ctx @ W1^T + b1) @ W2^T + b2.
  gemm_nt<1, true ><<<dim3(32, 64), 128, 0, stream>>>(ctxb, w1b, b1, nullptr, h1b, 4096, 2048, 1024);
  gemm_nt<0, false><<<dim3(16, 64), 128, 0, stream>>>(h1b, w2b, b2, out, nullptr, 4096, 1024, 2048);
}
